// SlidingWindowAttention_14104672600699
// MI455X (gfx1250) — matrix operand order repaired and checked by reading, not compiled, not run on hardware
//
#include <hip/hip_runtime.h>

typedef _Float16 half_t;
typedef __attribute__((ext_vector_type(16))) _Float16 v16h;
typedef __attribute__((ext_vector_type(8)))  float    v8f;

#define BATCH  2
#define SEQ    4096
#define DMODEL 1024
#define NHEAD  16
#define HDIM   64
#define WIN    256
#define NROWS  (BATCH * SEQ)   // 8192

// A-fragment (16x32 f16) K offset for element-pair g (0..7) in lane-half hs:
// VGPR0..7 hold K={0,1},{2,3}..{14,15} (+16 for lanes 16..31), contiguous.
__device__ __forceinline__ int a_koff(int g, int hs) {
    return 2 * g + 16 * hs;
}

// ---------------------------------------------------------------------------
// Projection GEMM:  out[n,m] = sum_k X[n,k] * Wt[m,k] + bias[m]
// MODE 0: epilogue = bias + interleaved RoPE, store f16 [B,H,S,HD]  (Q, K)
// MODE 1: epilogue = bias, store f16 transposed [B,H,HD,S]          (V)
// MODE 2: epilogue = bias, store f32 row-major [N, DMODEL]          (out proj)
// Workgroup tile: 128 rows x 64 feats; 8 waves, each 32x32 (4 WMMA tiles).
// Bandwidth-bound: rely on 192MB L2 for cross-wave reuse; prefetch next
// K-slice (global_prefetch_b8) to hide HBM latency behind the WMMAs.
// ---------------------------------------------------------------------------
template <int MODE>
__global__ __launch_bounds__(256) void proj_gemm(
    const float* __restrict__ X, const float* __restrict__ Wt,
    const float* __restrict__ bias, half_t* __restrict__ out16,
    float* __restrict__ out32)
{
    const int lane = threadIdx.x & 31;
    const int wave = threadIdx.x >> 5;
    const int hs   = lane >> 4;
    const int l16  = lane & 15;
    const int nbase = blockIdx.x * 128 + (wave >> 1) * 32;
    const int mbase = blockIdx.y * 64  + (wave & 1)  * 32;

    v8f c[2][2] = {};

    for (int kb = 0; kb < DMODEL; kb += 32) {
        v16h a[2], bf[2];
        #pragma unroll
        for (int i = 0; i < 2; ++i) {
            const float* rp = X + (size_t)(nbase + i * 16 + l16) * DMODEL + kb;
            if (kb + 32 < DMODEL) __builtin_prefetch(rp + 32, 0, 0);
            #pragma unroll
            for (int g = 0; g < 8; ++g) {
                int k = a_koff(g, hs);
                a[i][2 * g]     = (half_t)rp[k];
                a[i][2 * g + 1] = (half_t)rp[k + 1];
            }
        }
        #pragma unroll
        for (int j = 0; j < 2; ++j) {
            // B lane: column m = mbase+j*16+l16, K run = kb + hs*16 .. +16 (contiguous)
            const float* rp = Wt + (size_t)(mbase + j * 16 + l16) * DMODEL + kb + hs * 16;
            if (kb + 32 < DMODEL) __builtin_prefetch(rp + 32, 0, 0);
            #pragma unroll
            for (int e = 0; e < 16; ++e) bf[j][e] = (half_t)rp[e];
        }
        #pragma unroll
        for (int i = 0; i < 2; ++i)
            #pragma unroll
            for (int j = 0; j < 2; ++j)
                c[i][j] = __builtin_amdgcn_wmma_f32_16x16x32_f16(
                    false, a[i], false, bf[j], (short)0, c[i][j], false, false);
    }

    #pragma unroll
    for (int i = 0; i < 2; ++i) {
        #pragma unroll
        for (int j = 0; j < 2; ++j) {
            const int m  = mbase + j * 16 + l16;
            const float bv = bias[m];
            #pragma unroll
            for (int r = 0; r < 8; ++r) {
                const int row = nbase + i * 16 + r + 8 * hs;
                float val = c[i][j][r] + bv;
                if (MODE == 2) {
                    out32[(size_t)row * DMODEL + m] = val;
                } else {
                    const int b  = row >> 12;         // / SEQ
                    const int s  = row & (SEQ - 1);
                    const int h  = m >> 6;
                    const int hd = m & 63;
                    if (MODE == 0) {
                        // interleaved RoPE: pair partner feature m^1 lives in lane^1
                        float other = __shfl_xor(val, 1, 32);
                        float expo  = -(float)(hd & ~1) * (9.210340371976184f / 64.0f);
                        float inv   = __expf(expo);   // 10000^(-2*pair/64)
                        float ang   = (float)s * inv;
                        float sn, cs;
                        __sincosf(ang, &sn, &cs);
                        float o = (hd & 1) ? (val * cs + other * sn)
                                           : (val * cs - other * sn);
                        out16[((size_t)((b * NHEAD + h) * SEQ + s)) * HDIM + hd] = (half_t)o;
                    } else { // MODE 1: V transposed
                        out16[((size_t)((b * NHEAD + h) * HDIM + hd)) * SEQ + s] = (half_t)val;
                    }
                }
            }
        }
    }
}

// ---------------------------------------------------------------------------
// Sliding-window attention, flash-style per wave (16 queries / wave).
// Q,K f16 [B,H,S,HD]; V f16 [B,H,HD,S]; Y f32 [B,S,H,HD].
// ---------------------------------------------------------------------------
__global__ __launch_bounds__(256) void attn_kernel(
    const half_t* __restrict__ Q, const half_t* __restrict__ K,
    const half_t* __restrict__ Vt, float* __restrict__ Y)
{
    __shared__ _Float16 pbuf[8][16][32];   // per-wave P relayout buffer (8 KB)

    const int lane = threadIdx.x & 31;
    const int wave = threadIdx.x >> 5;
    const int hs   = lane >> 4;
    const int l16  = lane & 15;

    const int idx   = blockIdx.x;
    const int qc    = idx & 31;            // 32 chunks of 128 queries
    const int h     = (idx >> 5) & 15;
    const int b     = idx >> 9;
    const int qbase = qc * 128 + wave * 16;
    const int n     = qbase >> 8;          // window block index
    const int ql    = qbase & 255;         // query pos within block
    const int ksb   = n * WIN - WIN;       // cat-index -> seq offset

    const half_t* Qh = Q  + (size_t)((b * NHEAD + h) * SEQ)  * HDIM;
    const half_t* Kh = K  + (size_t)((b * NHEAD + h) * SEQ)  * HDIM;
    const half_t* Vh = Vt + (size_t)((b * NHEAD + h) * HDIM) * SEQ;

    // Q A-fragments (scale folded in)
    const float scale = 0.125f;            // 1/sqrt(64)
    v16h qa[2];
    #pragma unroll
    for (int t = 0; t < 2; ++t) {
        const half_t* qp = Qh + (size_t)(qbase + l16) * HDIM + t * 32;
        #pragma unroll
        for (int g = 0; g < 8; ++g) {
            int k = a_koff(g, hs);
            qa[t][2 * g]     = (half_t)(scale * (float)qp[k]);
            qa[t][2 * g + 1] = (half_t)(scale * (float)qp[k + 1]);
        }
    }

    v8f y[4] = {};
    float mrow[8], lrow[8];
    #pragma unroll
    for (int r = 0; r < 8; ++r) { mrow[r] = -1e30f; lrow[r] = 0.0f; }

    // only visit key chunks intersecting the band [i, i+WIN]
    int jc0 = ql >> 5;
    int jc1 = (ql + 15 + WIN) >> 5; if (jc1 > 15) jc1 = 15;

    for (int jc = jc0; jc <= jc1; ++jc) {
        const int kcat0 = jc * 32;

        // prefetch next chunk's K/V lines into L2/L0 while this chunk computes
        if (jc < jc1) {
            int pk = ksb + kcat0 + 32 + l16;  if (pk < 0) pk = 0;
            __builtin_prefetch(Kh + (size_t)pk * HDIM + hs * 32, 0, 0);
            int pv = ksb + kcat0 + 32 + hs * 16; if (pv < 0) pv = 0;
            __builtin_prefetch(Vh + (size_t)l16 * SEQ + pv, 0, 0);
        }

        v8f sf[2];
        #pragma unroll
        for (int jt = 0; jt < 2; ++jt) {
            const int kcat = kcat0 + jt * 16 + l16;
            int ks = ksb + kcat; int ksc = ks < 0 ? 0 : ks;  // clamp (masked anyway)
            v8f acc = {};
            #pragma unroll
            for (int t = 0; t < 2; ++t) {
                const half_t* kp = Kh + (size_t)ksc * HDIM + t * 32 + hs * 16;
                v16h kb;
                #pragma unroll
                for (int e = 0; e < 16; ++e) kb[e] = kp[e];
                acc = __builtin_amdgcn_wmma_f32_16x16x32_f16(
                    false, qa[t], false, kb, (short)0, acc, false, false);
            }
            #pragma unroll
            for (int r = 0; r < 8; ++r) {
                const int qloc = ql + r + 8 * hs;
                const int jcat = kcat0 + jt * 16 + l16;
                const bool valid = (jcat >= qloc) && (jcat <= qloc + WIN) &&
                                   ((n > 0) || (jcat >= WIN));
                sf[jt][r] = valid ? acc[r] : -3.0e38f;
            }
        }

        // online softmax per query row
        #pragma unroll
        for (int r = 0; r < 8; ++r) {
            float cm = fmaxf(sf[0][r], sf[1][r]);
            #pragma unroll
            for (int off = 1; off < 16; off <<= 1)
                cm = fmaxf(cm, __shfl_xor(cm, off, 32));
            float nm    = fmaxf(mrow[r], cm);
            float alpha = __expf(mrow[r] - nm);
            mrow[r] = nm;
            float p0 = __expf(sf[0][r] - nm);
            float p1 = __expf(sf[1][r] - nm);
            sf[0][r] = p0; sf[1][r] = p1;
            float rs = p0 + p1;
            #pragma unroll
            for (int off = 1; off < 16; off <<= 1)
                rs += __shfl_xor(rs, off, 32);
            lrow[r] = lrow[r] * alpha + rs;
            #pragma unroll
            for (int ht = 0; ht < 4; ++ht) y[ht][r] *= alpha;
        }

        // relayout P: D-layout -> LDS -> A-layout (in-order LDS within a wave)
        #pragma unroll
        for (int jt = 0; jt < 2; ++jt)
            #pragma unroll
            for (int r = 0; r < 8; ++r)
                pbuf[wave][r + 8 * hs][jt * 16 + l16] = (half_t)sf[jt][r];
        __builtin_amdgcn_wave_barrier();
        v16h pa;
        #pragma unroll
        for (int g = 0; g < 8; ++g) {
            int k = a_koff(g, hs);
            pa[2 * g]     = pbuf[wave][l16][k];
            pa[2 * g + 1] = pbuf[wave][l16][k + 1];
        }

        // Y += P @ V   (V transposed: contiguous 16-key runs per lane)
        int vs = ksb + kcat0 + hs * 16; int vsc = vs < 0 ? 0 : vs;
        #pragma unroll
        for (int ht = 0; ht < 4; ++ht) {
            const half_t* vp = Vh + (size_t)(ht * 16 + l16) * SEQ + vsc;
            v16h vb;
            #pragma unroll
            for (int e = 0; e < 16; ++e) vb[e] = vp[e];
            y[ht] = __builtin_amdgcn_wmma_f32_16x16x32_f16(
                false, pa, false, vb, (short)0, y[ht], false, false);
        }
    }

    // normalize and store f32 [B,S,H,HD]
    #pragma unroll
    for (int ht = 0; ht < 4; ++ht)
        #pragma unroll
        for (int r = 0; r < 8; ++r) {
            const int q  = qbase + r + 8 * hs;
            const int hd = ht * 16 + l16;
            Y[((size_t)((b * SEQ + q) * NHEAD + h)) * HDIM + hd] = y[ht][r] / lrow[r];
        }
}

// ---------------------------------------------------------------------------
extern "C" void kernel_launch(void* const* d_in, const int* in_sizes, int n_in,
                              void* d_out, int out_size, void* d_ws, size_t ws_size,
                              hipStream_t stream)
{
    const float* x    = (const float*)d_in[0];
    const float* wq_w = (const float*)d_in[1];
    const float* wq_b = (const float*)d_in[2];
    const float* wk_w = (const float*)d_in[3];
    const float* wk_b = (const float*)d_in[4];
    const float* wv_w = (const float*)d_in[5];
    const float* wv_b = (const float*)d_in[6];
    const float* wo_w = (const float*)d_in[7];
    const float* wo_b = (const float*)d_in[8];
    float* out = (float*)d_out;

    // workspace layout: Qf16 | Kf16 | Vf16(transposed) | Y f32   (~80 MB)
    const size_t nqkv = (size_t)BATCH * NHEAD * SEQ * HDIM;  // 8,388,608
    half_t* Qh = (half_t*)d_ws;
    half_t* Kh = Qh + nqkv;
    half_t* Vh = Kh + nqkv;
    float*  Yw = (float*)(Vh + nqkv);

    dim3 gg(NROWS / 128, DMODEL / 64);
    dim3 bb(256);
    proj_gemm<0><<<gg, bb, 0, stream>>>(x, wq_w, wq_b, Qh, nullptr);
    proj_gemm<0><<<gg, bb, 0, stream>>>(x, wk_w, wk_b, Kh, nullptr);
    proj_gemm<1><<<gg, bb, 0, stream>>>(x, wv_w, wv_b, Vh, nullptr);

    attn_kernel<<<dim3(BATCH * NHEAD * (SEQ / 128)), bb, 0, stream>>>(Qh, Kh, Vh, Yw);

    proj_gemm<2><<<gg, bb, 0, stream>>>(Yw, wo_w, wo_b, nullptr, out);
}
